// Criterion_47493748359597
// MI455X (gfx1250) — matrix operand order repaired and checked by reading, not compiled, not run on hardware
//
#include <hip/hip_runtime.h>
#include <hip/hip_bf16.h>

typedef __attribute__((ext_vector_type(16))) _Float16 v16h;
typedef __attribute__((ext_vector_type(8)))  _Float16 v8h;
typedef __attribute__((ext_vector_type(8)))  float    v8f;

#define NBINS 51
#define D_DIM 128
#define BS    1024
#define BW    (2.0f / (NBINS - 1))
#define INV_BW ((NBINS - 1) * 0.5f)

// ---------------------------------------------------------------------------
// Kernel 0: convert x f32 -> f16 workspace; block 0 zeroes global histograms.
// ---------------------------------------------------------------------------
__global__ __launch_bounds__(256) void k_convert(const float* __restrict__ x,
                                                 _Float16* __restrict__ xh,
                                                 float* __restrict__ gh) {
  int idx = blockIdx.x * 256 + threadIdx.x;
  if (idx < BS * D_DIM) xh[idx] = (_Float16)x[idx];
  if (blockIdx.x == 0 && threadIdx.x < 2 * NBINS) gh[threadIdx.x] = 0.0f;
}

// ---------------------------------------------------------------------------
// Kernel 1: fused sim = x x^T (WMMA f16->f32) + soft-binned pair histograms.
// Each block: 64x64 tile of sim. 8 waves; wave w -> row-tile w/2, col-tiles
// 2*(w&1) and 2*(w&1)+1. Only upper-triangular blocks do work.
// gh layout: [0..50] = pos histogram, [51..101] = neg histogram (unnormalized)
// ---------------------------------------------------------------------------
__global__ __launch_bounds__(256) void k_gemm_hist(const _Float16* __restrict__ xh,
                                                   const int* __restrict__ labels,
                                                   float* __restrict__ gh) {
  const int bi = blockIdx.y;   // block row  (i / 64)
  const int bj = blockIdx.x;   // block col  (j / 64)
  if (bj < bi) return;         // strictly lower 64x64 blocks have no i<j pairs

  __shared__ float hloc[2 * NBINS];
  __shared__ int   lrow[64];
  __shared__ int   lcol[64];

  const int tid = threadIdx.x;
  if (tid < 2 * NBINS) hloc[tid] = 0.0f;
  if (tid < 64)        lrow[tid] = labels[bi * 64 + tid];
  else if (tid < 128)  lcol[tid - 64] = labels[bj * 64 + (tid - 64)];
  __syncthreads();

  const int wave = tid >> 5;       // 0..7  (wave32)
  const int lane = tid & 31;
  const int sel  = lane >> 4;      // half-wave select
  const int ln   = lane & 15;

  const int tr     = wave >> 1;            // 0..3 : 16-row tile in block
  const int tcBase = (wave & 1) * 2;       // 0 or 2: first of two 16-col tiles

  const int rowBase = bi * 64 + tr * 16;

#pragma unroll
  for (int tci = 0; tci < 2; ++tci) {
    const int tc      = tcBase + tci;
    const int colBase = bj * 64 + tc * 16;

    v8f acc = {};
#pragma unroll
    for (int kb = 0; kb < 4; ++kb) {
      // A fragment: 16x32 f16. Lane ln = row M; sel picks the interleaved
      // K-halves per ISA layout (halves 0-7: K = kb*32+sel*8.., 8-15: +16).
      const _Float16* pa = xh + (size_t)(rowBase + ln) * D_DIM + kb * 32;
      v16h a;
      ((v8h*)&a)[0] = *(const v8h*)(pa + sel * 8);
      ((v8h*)&a)[1] = *(const v8h*)(pa + 16 + sel * 8);

      // B fragment: 32x16 f16, B[k][n] = x[colBase+n][k]. Lane ln = col N;
      // same interleaved K order as A (halves 0-7: K = kb*32+sel*8.., 8-15: +16).
      const _Float16* pb = xh + (size_t)(colBase + ln) * D_DIM + kb * 32;
      v16h b;
      ((v8h*)&b)[0] = *(const v8h*)(pb + sel * 8);
      ((v8h*)&b)[1] = *(const v8h*)(pb + 16 + sel * 8);

      acc = __builtin_amdgcn_wmma_f32_16x16x32_f16(
          /*neg_a=*/false, a, /*neg_b=*/false, b,
          /*c_mod=*/(short)0, acc, /*reuse_a=*/false, /*reuse_b=*/false);
    }

    // C layout: VGPR r, half-wave sel -> (M = r + 8*sel, N = ln).
    const int j    = colBase + ln;
    const int labj = lcol[tc * 16 + ln];
    const int mBase = tr * 16 + 8 * sel;            // row offset within block
#pragma unroll
    for (int r = 0; r < 8; ++r) {
      const int i = bi * 64 + mBase + r;
      if (i < j) {
        const float s = acc[r];
        int   b   = (int)floorf((s + 1.0f) * INV_BW);
        b         = min(max(b, 0), NBINS - 1);
        const float v    = (float)b * BW - 1.0f;
        const float w_hi = (s - v) * INV_BW;
        const float w_lo = 1.0f - w_hi;             // (v + bw - s)/bw
        const int   bh   = min(b + 1, NBINS - 1);
        const int   off  = (lrow[mBase + r] == labj) ? 0 : NBINS;
        atomicAdd(&hloc[off + b],  w_lo);           // ds_add_f32
        atomicAdd(&hloc[off + bh], w_hi);
      }
    }
  }

  __syncthreads();
  if (tid < 2 * NBINS) atomicAdd(&gh[tid], hloc[tid]);  // global_atomic_add_f32
}

// ---------------------------------------------------------------------------
// Kernel 2: normalize, cumsum(pos), loss = sum(neg_norm * cdf_pos). 51 bins.
// ---------------------------------------------------------------------------
__global__ void k_finalize(const float* __restrict__ gh, float* __restrict__ out) {
  if (threadIdx.x == 0 && blockIdx.x == 0) {
    float sp = 0.0f, sn = 0.0f;
    for (int k = 0; k < NBINS; ++k) { sp += gh[k]; sn += gh[NBINS + k]; }
    const float invp = (sp > 0.0f) ? 1.0f / sp : 0.0f;
    const float invn = (sn > 0.0f) ? 1.0f / sn : 0.0f;
    float cdf = 0.0f, loss = 0.0f;
    for (int k = 0; k < NBINS; ++k) {
      cdf  += gh[k] * invp;
      loss += gh[NBINS + k] * invn * cdf;
    }
    out[0] = loss;
  }
}

// ---------------------------------------------------------------------------
extern "C" void kernel_launch(void* const* d_in, const int* in_sizes, int n_in,
                              void* d_out, int out_size, void* d_ws, size_t ws_size,
                              hipStream_t stream) {
  const float* x      = (const float*)d_in[0];
  const int*   labels = (const int*)d_in[1];
  float*       out    = (float*)d_out;

  // Workspace layout: [0, 256KB) f16 copy of x; then 2*NBINS float histograms.
  _Float16* xh = (_Float16*)d_ws;
  float*    gh = (float*)((char*)d_ws + (size_t)BS * D_DIM * sizeof(_Float16));

  k_convert<<<dim3((BS * D_DIM) / 256), dim3(256), 0, stream>>>(x, xh, gh);

  dim3 grid(BS / 64, BS / 64);   // 16 x 16; lower-triangular blocks early-exit
  k_gemm_hist<<<grid, dim3(256), 0, stream>>>(xh, labels, gh);

  k_finalize<<<dim3(1), dim3(64), 0, stream>>>(gh, out);
}
